// TripletLoss_41609643163994
// MI455X (gfx1250) — matrix operand order repaired and checked by reading, not compiled, not run on hardware
//
#include <hip/hip_runtime.h>
#include <hip/hip_bf16.h>

#define N_ANCH 8192
#define DIM    1024
#define MARGIN 0.2f
#define EPS_P  1e-6f
#define WAVES  4          // waves per block in the distance kernel
#define KCHUNK 32         // K per WMMA
#define NCHUNK (DIM / KCHUNK)

typedef __attribute__((ext_vector_type(16))) _Float16 v16h;
typedef __attribute__((ext_vector_type(8)))  float    v8f;

union V16H { v16h v; _Float16 e[16]; };

// ---------------------------------------------------------------------------
// Counter-based hash: monotone stand-in for the per-(i,j) Gumbel noise.
// argmax over a row of hashes == uniform masked sample (Gumbel-max trick).
// ---------------------------------------------------------------------------
__device__ __forceinline__ unsigned hash2d(unsigned i, unsigned j) {
    unsigned h = i * 0x9E3779B1u + j * 0x85EBCA77u + 0x165667B1u;
    h ^= h >> 15; h *= 0x2C1B3C6Du;
    h ^= h >> 12; h *= 0x297A2D39u;
    h ^= h >> 15;
    return h;
}

// ---------------------------------------------------------------------------
// Kernel 1: per-anchor masked argmax over the hash row.
// One 256-thread block per anchor; categories staged in LDS (32 KB).
// ---------------------------------------------------------------------------
__global__ void triplet_sample_kernel(const int* __restrict__ cat,
                                      int* __restrict__ pos_idx,
                                      int* __restrict__ neg_idx,
                                      int* __restrict__ valid) {
    __shared__ int s_cat[N_ANCH];
    __shared__ unsigned long long skey[2][256];
    __shared__ int sidx[2][256];

    const int i   = blockIdx.x;
    const int tid = threadIdx.x;

    for (int j = tid; j < N_ANCH; j += blockDim.x) s_cat[j] = cat[j];
    __syncthreads();

    const int ci = s_cat[i];
    unsigned long long bkp = 0ull, bkn = 0ull;
    int bip = -1, bin = -1;

    for (int j = tid; j < N_ANCH; j += blockDim.x) {
        const unsigned h = hash2d((unsigned)i, (unsigned)j);
        // tie-break toward lower j (argmax-first semantics)
        const unsigned long long key =
            ((unsigned long long)h << 20) | (unsigned)(N_ANCH - 1 - j);
        const bool same = (s_cat[j] == ci);
        if (same) {
            if (j != i && key > bkp) { bkp = key; bip = j; }
        } else {
            if (key > bkn) { bkn = key; bin = j; }
        }
    }

    skey[0][tid] = bkp; sidx[0][tid] = bip;
    skey[1][tid] = bkn; sidx[1][tid] = bin;
    __syncthreads();

    for (int s = blockDim.x >> 1; s > 0; s >>= 1) {
        if (tid < s) {
            #pragma unroll
            for (int t = 0; t < 2; ++t) {
                if (skey[t][tid + s] > skey[t][tid]) {
                    skey[t][tid] = skey[t][tid + s];
                    sidx[t][tid] = sidx[t][tid + s];
                }
            }
        }
        __syncthreads();
    }

    if (tid == 0) {
        const int p = sidx[0][0], n = sidx[1][0];
        pos_idx[i] = (p < 0) ? 0 : p;
        neg_idx[i] = (n < 0) ? 0 : n;
        valid[i]   = (p >= 0 && n >= 0) ? 1 : 0;
    }
}

// ---------------------------------------------------------------------------
// Kernel 2: pairwise distances via WMMA Gram diagonal.
// One wave32 per 16-anchor tile; D=1024 swept in 32 chunks of K=32.
// diff rows are built in the exact A (16x32) / B (32x16) f16 VGPR striping
// from CDNA5 ISA, accumulated with v_wmma_f32_16x16x32_f16.
// ---------------------------------------------------------------------------
__global__ void __launch_bounds__(32 * WAVES)
triplet_dist_kernel(const float* __restrict__ eeg,
                    const float* __restrict__ text,
                    const int*   __restrict__ pos_idx,
                    const int*   __restrict__ neg_idx,
                    const int*   __restrict__ valid,
                    float* __restrict__ loss) {
    __shared__ float sdiag[WAVES][2][16];

    const int lane = threadIdx.x & 31;
    const int wave = threadIdx.x >> 5;
    const int tile = blockIdx.x * WAVES + wave;   // 512 tiles of 16 anchors

    const int hh   = lane >> 4;    // half-wave select (ISA lane group)
    const int mrow = lane & 15;    // row M for A-operand, col N for B-operand

    const int anchor_row = tile * 16 + mrow;
    const int pidx = pos_idx[anchor_row];
    const int nidx = neg_idx[anchor_row];

    const float* __restrict__ Arow = eeg  + (size_t)anchor_row * DIM;
    const float* __restrict__ Prow = text + (size_t)pidx       * DIM;
    const float* __restrict__ Nrow = text + (size_t)nidx       * DIM;

    v8f cp = {};   // pos Gram accumulator (16x16 f32)
    v8f cn = {};   // neg Gram accumulator

    for (int kc = 0; kc < NCHUNK; ++kc) {
        const int k0 = kc * KCHUNK;

        if (kc + 1 < NCHUNK) {            // gfx1250 global_prefetch_b8 path
            __builtin_prefetch(Arow + k0 + KCHUNK, 0, 1);
            __builtin_prefetch(Prow + k0 + KCHUNK, 0, 1);
            __builtin_prefetch(Nrow + k0 + KCHUNK, 0, 1);
        }

        // --- gather this lane's K elements in A-layout and B-layout ---
        float aA[16], pA[16], nA[16];     // A operand: K = ((h<8)?h:h+8) + hh*8
        float aB[16], pB[16], nB[16];     // B operand: K = ((h<8)?h:h+8) + hh*8
        #pragma unroll
        for (int h = 0; h < 16; ++h) {
            const int kA = k0 + ((h < 8) ? h : (h + 8)) + hh * 8;
            const int kB = k0 + ((h < 8) ? h : (h + 8)) + hh * 8;
            aA[h] = Arow[kA];  pA[h] = Prow[kA];  nA[h] = Nrow[kA];
            aB[h] = Arow[kB];  pB[h] = Prow[kB];  nB[h] = Nrow[kB];
        }

        V16H Ap, An, Bp, Bn;
        #pragma unroll
        for (int h = 0; h < 16; ++h) {
            Ap.e[h] = (_Float16)(aA[h] - pA[h] + EPS_P);
            An.e[h] = (_Float16)(aA[h] - nA[h] + EPS_P);
            Bp.e[h] = (_Float16)(aB[h] - pB[h] + EPS_P);
            Bn.e[h] = (_Float16)(aB[h] - nB[h] + EPS_P);
        }

        // D = A x B + C  (uniform EXEC across the wave — no divergence here)
        cp = __builtin_amdgcn_wmma_f32_16x16x32_f16(
                 false, Ap.v, false, Bp.v, (short)0, cp, false, false);
        cn = __builtin_amdgcn_wmma_f32_16x16x32_f16(
                 false, An.v, false, Bn.v, (short)0, cn, false, false);
    }

    // --- extract Gram diagonal: row M lives at lane (M<8?M:M+16), VGPR M%8 ---
    #pragma unroll
    for (int r = 0; r < 8; ++r) {
        const int M = r + hh * 8;
        if (mrow == M) {
            sdiag[wave][0][M] = cp[r];
            sdiag[wave][1][M] = cn[r];
        }
    }
    __syncthreads();

    if (lane < 16) {
        const int i = tile * 16 + lane;
        const float pd = sqrtf(fmaxf(sdiag[wave][0][lane], 0.0f));
        const float nd = sqrtf(fmaxf(sdiag[wave][1][lane], 0.0f));
        const float l  = fmaxf(pd - nd + MARGIN, 0.0f);
        loss[i] = valid[i] ? l : 0.0f;
    }
}

// ---------------------------------------------------------------------------
// Kernel 3: final reduction -> scalar loss.
// ---------------------------------------------------------------------------
__global__ void triplet_reduce_kernel(const float* __restrict__ loss,
                                      const int*   __restrict__ valid,
                                      float* __restrict__ out) {
    __shared__ float ss[256];
    __shared__ int   sc[256];
    const int tid = threadIdx.x;

    float s = 0.0f; int c = 0;
    for (int j = tid; j < N_ANCH; j += 256) { s += loss[j]; c += valid[j]; }
    ss[tid] = s; sc[tid] = c;
    __syncthreads();

    for (int st = 128; st > 0; st >>= 1) {
        if (tid < st) { ss[tid] += ss[tid + st]; sc[tid] += sc[tid + st]; }
        __syncthreads();
    }
    if (tid == 0) out[0] = (sc[0] > 0) ? (ss[0] / (float)sc[0]) : 0.0f;
}

// ---------------------------------------------------------------------------
extern "C" void kernel_launch(void* const* d_in, const int* in_sizes, int n_in,
                              void* d_out, int out_size, void* d_ws, size_t ws_size,
                              hipStream_t stream) {
    (void)in_sizes; (void)n_in; (void)out_size; (void)ws_size;

    const float* eeg  = (const float*)d_in[0];
    const float* text = (const float*)d_in[1];
    const int*   cats = (const int*)d_in[2];
    float*       out  = (float*)d_out;

    int*   pos  = (int*)d_ws;
    int*   neg  = pos + N_ANCH;
    int*   val  = neg + N_ANCH;
    float* loss = (float*)(val + N_ANCH);

    triplet_sample_kernel<<<N_ANCH, 256, 0, stream>>>(cats, pos, neg, val);
    triplet_dist_kernel<<<N_ANCH / 16 / WAVES, 32 * WAVES, 0, stream>>>(
        eeg, text, pos, neg, val, loss);
    triplet_reduce_kernel<<<1, 256, 0, stream>>>(loss, val, out);
}
